// PeriodicityTransform_36309653521021
// MI455X (gfx1250) — matrix operand order repaired and checked by reading, not compiled, not run on hardware
//
#include <hip/hip_runtime.h>

// ---------------------------------------------------------------------------
// PeriodicityTransform for MI455X (gfx1250, wave32).
//
// DFT-as-GEMM via v_wmma_f32_16x16x32_f16 with hi/lo f16 split-precision
// compensation. Trig basis: one 8-byte LDS record per phase (cos/sin, hi/lo
// packed) gathered with ds_load_b64, transposed into WMMA B fragments with
// v_perm_b32 (one per destination VGPR). The f16 sequence tile is stored
// pre-swizzled so a WMMA A fragment is one aligned 32-byte LDS load.
// Fused: spectrum -> top-4 -> gather/mask emission with non-temporal stores.
// ---------------------------------------------------------------------------

typedef __attribute__((ext_vector_type(16))) _Float16     v16h;
typedef __attribute__((ext_vector_type(8)))  float        v8f;
typedef __attribute__((ext_vector_type(4)))  float        v4f;
typedef __attribute__((ext_vector_type(8)))  unsigned int v8u;
typedef __attribute__((ext_vector_type(2)))  unsigned int v2u;

#define T_LEN 512
#define N_CH  128
#define B_SZ  32
#define KTOP  4
#define PMAXD 64
#define MINP  16
#define CMAX  32                 // T / MINP
#define ROWS  16                 // sequences handled per workgroup
#define NFREQ 256                // rfft bins 1..256 (DC is zeroed by reference)
#define SEQ_STRIDE  516          // f32 row stride (pad -> conflict-free)
#define SEQH_STRIDE 528          // f16 row stride (mult of 16 -> 32B aligned)
#define AMP_STRIDE  260

__device__ __forceinline__ v8f wmma_f16(v16h a, v16h b, v8f c) {
  // (neg_a, A, neg_b, B, c_mod, C, reuse_a, reuse_b)
  return __builtin_amdgcn_wmma_f32_16x16x32_f16(false, a, false, b,
                                                (short)0, c, false, false);
}

__device__ __forceinline__ unsigned short h_bits(_Float16 h) {
  return __builtin_bit_cast(unsigned short, h);
}

__global__ __launch_bounds__(256)
void periodicity_wmma_kernel(const float* __restrict__ x,
                             float* __restrict__ gathered,
                             float* __restrict__ maskout,
                             float* __restrict__ kampout) {
  __shared__ float    lds_seq [ROWS * SEQ_STRIDE];   // f32 tile (for gather)
  __shared__ _Float16 lds_seqh[ROWS * SEQH_STRIDE];  // f16 hi tile, swizzled
  __shared__ _Float16 lds_seql[ROWS * SEQH_STRIDE];  // f16 lo tile, swizzled
  // Per-phase record: .x = cos_hi | cos_lo<<16 ; .y = sin_hi | sin_lo<<16
  __shared__ v2u      lds_trig[T_LEN];
  __shared__ float    lds_amp [ROWS * AMP_STRIDE];
  __shared__ float    rk_amp [ROWS * KTOP];
  __shared__ int      rk_P   [ROWS * KTOP];
  __shared__ int      rk_cyc [ROWS * KTOP];
  __shared__ int      rk_base[ROWS * KTOP];

  const int tid    = threadIdx.x;
  const int lane   = tid & 31;
  const int wave   = tid >> 5;          // 8 waves
  const int laneLo = lane & 15;
  const int laneHi = lane >> 4;         // 0 / 1
  const int bn0    = blockIdx.x * ROWS; // first sequence of this tile
  const int b      = bn0 / N_CH;
  const int n0     = bn0 % N_CH;

  // ---- Phase 0: stage seqs tile into LDS (f32 + swizzled f16 hi/lo) -------
  // x is [B, T, N]; seqs[bn][t] = x[b][t][n].
  // f16 swizzle per 32-wide chunk: K order [0-7, 16-23, 8-15, 24-31], so a
  // lane's 16 A-fragment halves (ISA 16-bit A 16x32 layout) are one
  // contiguous, 32-byte-aligned run: lanes<16 -> pos 0..15, lanes>=16 -> 16..31.
  {
    const int j  = tid & 15;            // channel within tile
    const int t0 = tid >> 4;            // 16 time rows per pass
    const float* src = x + (size_t)b * T_LEN * N_CH + n0 + j;
    for (int t = t0; t < T_LEN; t += 16) {
      const float v = src[(size_t)t * N_CH];
      lds_seq[j * SEQ_STRIDE + t] = v;
      const _Float16 hi = (_Float16)v;
      const int k   = t & 31;
      const int pos = (k & 7) | ((k & 16) >> 1) | ((k & 8) << 1);
      const int hidx = j * SEQH_STRIDE + (t & ~31) + pos;
      lds_seqh[hidx] = hi;
      lds_seql[hidx] = (_Float16)(v - (float)hi);
    }
    // One-time trig records (512 phases).
    for (int ph = tid; ph < T_LEN; ph += 256) {
      const float theta = (float)ph * 0.01227184630308513f;  // 2*pi/512
      float s, c;
      __sincosf(theta, &s, &c);
      const _Float16 chh = (_Float16)c;
      const _Float16 shh = (_Float16)s;
      const _Float16 cll = (_Float16)(c - (float)chh);
      const _Float16 sll = (_Float16)(s - (float)shh);
      v2u r;
      r.x = (unsigned int)h_bits(chh) | ((unsigned int)h_bits(cll) << 16);
      r.y = (unsigned int)h_bits(shh) | ((unsigned int)h_bits(sll) << 16);
      lds_trig[ph] = r;
    }
  }
  __syncthreads();

  // ---- Phase 1: spectrum via WMMA GEMM -------------------------------------
  // Re[16 rows, 256 bins] = seqs @ cos,  Im = seqs @ sin, amp = sqrt(Re^2+Im^2)
  // Split-precision: A*B ~= Ah*Bh + Ah*Bl + Al*Bh  (err ~2^-22 -> f32-grade).
  for (int nt2 = 0; nt2 < 2; ++nt2) {
    const int ntile = wave * 2 + nt2;              // 16 bins per tile, 16 tiles
    const int freq  = ntile * 16 + laneLo + 1;     // this lane's B column bin
    v8f re = {};
    v8f im = {};
    for (int kc = 0; kc < 16; ++kc) {
      const int t0k = kc * 32;
      // A fragments: single aligned 32-byte LDS load each (swizzled layout).
      const int abase = laneLo * SEQH_STRIDE + t0k + laneHi * 16;
      const v16h ah = *(const v16h*)&lds_seqh[abase];
      const v16h al = *(const v16h*)&lds_seql[abase];
      // B fragments: 16 ds_load_b64 gathers, then v_perm_b32 transpose.
      // B 32x16 layout: lane column N = lane%16; half h -> K = (h&7) + 8*laneHi + 2*(h&8).
      unsigned int cw[16], sw[16];
      const int tbase = t0k + 8 * laneHi;
#pragma unroll
      for (int h = 0; h < 16; ++h) {
        const int ph = (freq * (tbase + (h & 7) + ((h & 8) << 1))) & (T_LEN - 1);  // exact int phase
        const v2u r = lds_trig[ph];
        cw[h] = r.x;
        sw[h] = r.y;
      }
      v8u bchw, bclw, bshw, bslw;
#pragma unroll
      for (int i = 0; i < 8; ++i) {
        // dest word i = halves (2i, 2i+1): lo16s -> hi table, hi16s -> lo table
        bchw[i] = __builtin_amdgcn_perm(cw[2*i+1], cw[2*i], 0x05040100u);
        bclw[i] = __builtin_amdgcn_perm(cw[2*i+1], cw[2*i], 0x07060302u);
        bshw[i] = __builtin_amdgcn_perm(sw[2*i+1], sw[2*i], 0x05040100u);
        bslw[i] = __builtin_amdgcn_perm(sw[2*i+1], sw[2*i], 0x07060302u);
      }
      const v16h bch = __builtin_bit_cast(v16h, bchw);
      const v16h bcl = __builtin_bit_cast(v16h, bclw);
      const v16h bsh = __builtin_bit_cast(v16h, bshw);
      const v16h bsl = __builtin_bit_cast(v16h, bslw);
      re = wmma_f16(ah, bch, re);
      re = wmma_f16(ah, bcl, re);
      re = wmma_f16(al, bch, re);
      im = wmma_f16(ah, bsh, im);
      im = wmma_f16(ah, bsl, im);
      im = wmma_f16(al, bsh, im);
    }
    // C/D layout: VGPR r, lanes 0-15 -> M=r, N=lane; lanes 16-31 -> M=r+8.
#pragma unroll
    for (int r = 0; r < 8; ++r) {
      const int row = r + 8 * laneHi;
      const int col = ntile * 16 + laneLo;         // bin-1
      lds_amp[row * AMP_STRIDE + col] =
          __builtin_sqrtf(re[r] * re[r] + im[r] * im[r]);
    }
  }
  __syncthreads();

  // ---- Phase 2: per-row top-4 (wave-parallel argmax, ties -> lowest idx) ---
  for (int rr = 0; rr < 2; ++rr) {
    const int row = wave * 2 + rr;                 // 8 waves x 2 rows = 16
    for (int kk = 0; kk < KTOP; ++kk) {
      float bv = -1.0f;
      int   bi = 0;
      for (int f = lane; f < NFREQ; f += 32) {
        const float v = lds_amp[row * AMP_STRIDE + f];
        if (v > bv) { bv = v; bi = f; }
      }
#pragma unroll
      for (int off = 16; off > 0; off >>= 1) {
        const float ov = __shfl_down(bv, off, 32);
        const int   oi = __shfl_down(bi, off, 32);
        if (ov > bv || (ov == bv && oi < bi)) { bv = ov; bi = oi; }
      }
      if (lane == 0) {
        lds_amp[row * AMP_STRIDE + bi] = -1e30f;   // remove winner
        const int kidx = bi + 1;                   // rfft bin index (>=1)
        int P = T_LEN / kidx;
        P = (P < MINP) ? MINP : ((P > PMAXD) ? PMAXD : P);
        const int cyc = T_LEN / P;                 // >= 8, <= 32
        rk_amp [row * KTOP + kk] = bv;
        rk_P   [row * KTOP + kk] = P;
        rk_cyc [row * KTOP + kk] = cyc;
        rk_base[row * KTOP + kk] = T_LEN - cyc * P;
      }
      // Same-wave LDS ops are in-order; this wave exclusively owns the row,
      // so the removal is visible to the next scan without a block barrier.
    }
  }
  __syncthreads();

  // ---- Phase 3: stream gather + mask with non-temporal 128-bit stores ------
  // Element space per WG: 16 rows x 4 k x 32 cycles x 16 float4-groups.
  for (int i = tid; i < ROWS * KTOP * CMAX * (PMAXD / 4); i += 256) {
    const int p4  = i & 15;
    const int c   = (i >> 4) & 31;
    const int kk  = (i >> 9) & 3;
    const int row = i >> 11;
    const int P    = rk_P   [row * KTOP + kk];
    const int cyc  = rk_cyc [row * KTOP + kk];
    const int base = rk_base[row * KTOP + kk];
    v4f g, m;
#pragma unroll
    for (int jj = 0; jj < 4; ++jj) {
      const int p   = p4 * 4 + jj;
      const bool ok = (c < cyc) && (p < P);        // in-range => idx < T always
      m[jj] = ok ? 1.0f : 0.0f;
      g[jj] = ok ? lds_seq[row * SEQ_STRIDE + base + c * P + p] : 0.0f;
    }
    const size_t off =
        (((size_t)(bn0 + row) * KTOP + kk) * CMAX + c) * PMAXD + p4 * 4;
    __builtin_nontemporal_store(g, (v4f*)(gathered + off));
    __builtin_nontemporal_store(m, (v4f*)(maskout + off));
  }

  // kamp: [BN, K] row-major == reshape(B, N, K)
  if (tid < ROWS * KTOP)
    kampout[(size_t)bn0 * KTOP + tid] = rk_amp[tid];
}

extern "C" void kernel_launch(void* const* d_in, const int* in_sizes, int n_in,
                              void* d_out, int out_size, void* d_ws, size_t ws_size,
                              hipStream_t stream) {
  (void)in_sizes; (void)n_in; (void)out_size; (void)d_ws; (void)ws_size;
  const float* x = (const float*)d_in[0];
  float* out = (float*)d_out;
  const size_t gsz = (size_t)B_SZ * N_CH * KTOP * CMAX * PMAXD; // 33,554,432
  float* gathered = out;
  float* maskp    = out + gsz;
  float* kampp    = out + 2 * gsz;
  dim3 grid((B_SZ * N_CH) / ROWS);   // 256 workgroups, one per 16 sequences
  dim3 block(256);                   // 8 wave32
  periodicity_wmma_kernel<<<grid, block, 0, stream>>>(x, gathered, maskp, kampp);
}
